// OriginalEndoMamba_52089363366669
// MI455X (gfx1250) — compile-verified, matrix operand order checked by reading, not run on hardware
//
#include <hip/hip_runtime.h>
#include <hip/hip_bf16.h>
#include <math.h>

// ---------------------------------------------------------------------------
// EndoMamba forward on MI455X (gfx1250): all GEMMs via v_wmma_f32_16x16x32_bf16
// ---------------------------------------------------------------------------

typedef __attribute__((ext_vector_type(16))) __bf16 v16bf;
typedef __attribute__((ext_vector_type(8)))  float  v8f;

#define NTOK   3152   // B*T*P = 2*8*197  (also B*(T*P) = 2*1576)
#define DMODEL 384
#define EMODEL 768
#define SSTATE 16
#define RRANK  24
#define PTOK   3136   // B*T*Np = 16*196

__device__ __forceinline__ float softplusf(float x) {
    return (x > 20.f) ? x : log1pf(__expf(x));
}

// Load a 16x32 bf16 fragment (A or B operand) for one matrix row `row`:
// lane-half h needs contiguous float runs [kb+8h, kb+8h+8) and [kb+16+8h, +8).
__device__ __forceinline__ v16bf load_frag(const float* __restrict__ base, int half) {
    const float4 r0 = *(const float4*)(base + half * 8);
    const float4 r1 = *(const float4*)(base + half * 8 + 4);
    const float4 r2 = *(const float4*)(base + 16 + half * 8);
    const float4 r3 = *(const float4*)(base + 16 + half * 8 + 4);
    v16bf f;
    f[0]  = (__bf16)r0.x; f[1]  = (__bf16)r0.y; f[2]  = (__bf16)r0.z; f[3]  = (__bf16)r0.w;
    f[4]  = (__bf16)r1.x; f[5]  = (__bf16)r1.y; f[6]  = (__bf16)r1.z; f[7]  = (__bf16)r1.w;
    f[8]  = (__bf16)r2.x; f[9]  = (__bf16)r2.y; f[10] = (__bf16)r2.z; f[11] = (__bf16)r2.w;
    f[12] = (__bf16)r3.x; f[13] = (__bf16)r3.y; f[14] = (__bf16)r3.z; f[15] = (__bf16)r3.w;
    return f;
}

// ------------------------- generic WMMA GEMM -------------------------------
// out[M,N] (f32, ldo) = A[M,K] (f32, lda) * W[N,K]^T (f32, ldw)  [+bias] [softplus]
// One wave computes a 16x32 tile (A fragment reused across 2 WMMAs).
// Requires: K % 32 == 0, lda/ldw multiples of 4 floats, M % 16 == 0 OK to clamp.
__global__ void gemm_wmma_kernel(const float* __restrict__ A, int lda,
                                 const float* __restrict__ W, int ldw,
                                 const float* __restrict__ bias,
                                 float* __restrict__ out, int ldo,
                                 int M, int N, int K, int act) {
    const int lane = threadIdx.x;                       // 0..31 (wave32)
    const int m0 = blockIdx.x * 16;
    const int n0 = (blockIdx.y * blockDim.y + threadIdx.y) * 32;
    if (m0 >= M || n0 >= N) return;

    const int r    = lane & 15;
    const int half = lane >> 4;
    const size_t arow  = (size_t)((m0 + r < M) ? (m0 + r) : (M - 1));
    const size_t wrow0 = (size_t)((n0 + r      < N) ? (n0 + r)      : (N - 1));
    const size_t wrow1 = (size_t)((n0 + 16 + r < N) ? (n0 + 16 + r) : (N - 1));

    v8f acc0 = {};
    v8f acc1 = {};
    const float* Ap = A + arow * lda;
    const float* Wp0 = W + wrow0 * ldw;
    const float* Wp1 = W + wrow1 * ldw;
    for (int kb = 0; kb < K; kb += 32) {
        const v16bf af  = load_frag(Ap  + kb, half);
        const v16bf bf0 = load_frag(Wp0 + kb, half);
        const v16bf bf1 = load_frag(Wp1 + kb, half);
        acc0 = __builtin_amdgcn_wmma_f32_16x16x32_bf16(
            false, af, false, bf0, (short)0, acc0, false, false);
        acc1 = __builtin_amdgcn_wmma_f32_16x16x32_bf16(
            false, af, false, bf1, (short)0, acc1, false, false);
    }

#pragma unroll
    for (int tile = 0; tile < 2; ++tile) {
        const int col = n0 + 16 * tile + r;
        if (col < N) {
            const v8f& acc = tile ? acc1 : acc0;
#pragma unroll
            for (int v = 0; v < 8; ++v) {
                const int row = m0 + v + 8 * half;       // C/D layout (ISA 7.12.2)
                if (row < M) {
                    float val = acc[v];
                    if (bias) val += bias[col];
                    if (act == 1) val = softplusf(val);
                    out[(size_t)row * ldo + col] = val;
                }
            }
        }
    }
}

// ------------------------- zero-pad dtw rows 24 -> 32 ----------------------
__global__ void pad_dtw_kernel(const float* __restrict__ dtw, float* __restrict__ out) {
    const int e = blockIdx.x;            // 0..767
    const int k = threadIdx.x;           // 0..31
    out[e * 32 + k] = (k < RRANK) ? dtw[e * RRANK + k] : 0.f;
}

// ------------------------- im2col for the patch conv -----------------------
__global__ void im2col_patch_kernel(const float* __restrict__ x, float* __restrict__ out) {
    const int row = blockIdx.x;            // 0..3135: (b,t,hp,wp)
    const int bt = row / 196, np = row % 196;
    const int b = bt >> 3, t = bt & 7;
    const int hp = np / 14, wp = np % 14;
    for (int j = 0; j < 3; ++j) {
        const int k = threadIdx.x + j * 256;   // 0..767 : c*256 + ph*16 + pw
        const int c = k >> 8, rem = k & 255;
        const int ph = rem >> 4, pw = rem & 15;
        const int h = hp * 16 + ph, w = wp * 16 + pw;
        const size_t ix = ((((size_t)b * 3 + c) * 8 + t) * 224 + h) * 224 + w;
        out[(size_t)row * 768 + k] = x[ix];
    }
}

// ------------------------- token assembly ----------------------------------
__global__ void assemble_kernel(const float* __restrict__ ptok, const float* __restrict__ cls,
                                const float* __restrict__ pos, float* __restrict__ hidden,
                                float* __restrict__ residual) {
    const int row = blockIdx.x;            // 0..3151 : (b,t,pp)
    const int bt = row / 197, pp = row % 197;
    const int t = bt & 7;
    for (int j = 0; j < 3; ++j) {
        const int d = threadIdx.x + j * 128;
        float v;
        if (pp == 0) {
            v = cls[d] + pos[d];
        } else {
            const int np = pp - 1;
            const int i2 = d & ~1;
            const float div = powf(10000.f, (float)i2 / (float)DMODEL);
            const float arg = (float)t / div;
            const float enc = (d & 1) ? cosf(arg) : sinf(arg);
            v = ptok[((size_t)bt * 196 + np) * DMODEL + d] + pos[(size_t)pp * DMODEL + d] + enc;
        }
        hidden[(size_t)row * DMODEL + d]   = v;
        residual[(size_t)row * DMODEL + d] = 0.f;
    }
}

// ------------------------- residual-add + RMSNorm --------------------------
__global__ void resid_rmsnorm_kernel(float* __restrict__ residual, const float* __restrict__ hidden,
                                     const float* __restrict__ w, float* __restrict__ out) {
    __shared__ float red[128];
    const int row = blockIdx.x;
    const int tid = threadIdx.x;
    float vals[3];
    float s = 0.f;
#pragma unroll
    for (int j = 0; j < 3; ++j) {
        const int d = tid + j * 128;
        const size_t ix = (size_t)row * DMODEL + d;
        const float v = residual[ix] + hidden[ix];
        residual[ix] = v;
        vals[j] = v;
        s += v * v;
    }
    red[tid] = s;
    __syncthreads();
    for (int st = 64; st > 0; st >>= 1) {
        if (tid < st) red[tid] += red[tid + st];
        __syncthreads();
    }
    const float scale = rsqrtf(red[0] / (float)DMODEL + 1e-5f);
#pragma unroll
    for (int j = 0; j < 3; ++j) {
        const int d = tid + j * 128;
        out[(size_t)row * DMODEL + d] = vals[j] * scale * w[d];
    }
}

// ------------------------- causal depthwise conv + SiLU --------------------
__global__ void conv_silu_kernel(const float* __restrict__ xz, const float* __restrict__ cw,
                                 const float* __restrict__ cb, float* __restrict__ u_act,
                                 int L, int dir) {
    const int row = blockIdx.x;
    const int seq = row / L, l = row % L;
    const size_t base = (size_t)seq * L;
    for (int j = 0; j < 3; ++j) {
        const int e = threadIdx.x + j * 256;
        float acc = cb[e];
#pragma unroll
        for (int k = 0; k < 4; ++k) {
            const int ll = l - 3 + k;
            if (ll >= 0) {
                const int pl = dir ? (L - 1 - ll) : ll;
                acc += cw[e * 4 + k] * xz[(base + pl) * (2 * EMODEL) + e];
            }
        }
        const float sg = 1.f / (1.f + __expf(-acc));
        u_act[(size_t)row * EMODEL + e] = acc * sg;
    }
}

// ------------------------- selective scan ----------------------------------
__global__ void __launch_bounds__(256) scan_kernel(
    const float* __restrict__ dbl, const float* __restrict__ dt,
    const float* __restrict__ u_act, const float* __restrict__ xz,
    const float* __restrict__ Alog, const float* __restrict__ Dp,
    float* __restrict__ y, int L, int dir, int accum) {
    __shared__ float sB[16], sC[16];
    const int seq = blockIdx.x, tid = threadIdx.x;
    const size_t base = (size_t)seq * L;

    float A[3][16], h[3][16], dcoef[3];
#pragma unroll
    for (int j = 0; j < 3; ++j) {
        const int e = tid + j * 256;
        dcoef[j] = Dp[e];
#pragma unroll
        for (int s = 0; s < 16; ++s) {
            A[j][s] = -__expf(Alog[e * 16 + s]);
            h[j][s] = 0.f;
        }
    }

    for (int tstep = 0; tstep < L; ++tstep) {
        const size_t row = base + tstep;
        if (tid < 16)      sB[tid]      = dbl[row * 64 + 24 + tid];   // B: cols 24..39
        else if (tid < 32) sC[tid - 16] = dbl[row * 64 + 24 + tid];   // C: cols 40..55
        __syncthreads();

        const size_t prow = base + (size_t)(dir ? (L - 1 - tstep) : tstep);
#pragma unroll
        for (int j = 0; j < 3; ++j) {
            const int e = tid + j * 256;
            const float dte = dt[row * EMODEL + e];
            const float ue  = u_act[row * EMODEL + e];
            const float du  = dte * ue;
            float acc = 0.f;
#pragma unroll
            for (int s = 0; s < 16; ++s) {
                const float hv = __expf(dte * A[j][s]) * h[j][s] + du * sB[s];
                h[j][s] = hv;
                acc += hv * sC[s];
            }
            const float z = xz[prow * (2 * EMODEL) + EMODEL + e];
            const float outv = (acc + ue * dcoef[j]) * (z / (1.f + __expf(-z)));
            const size_t oix = prow * EMODEL + e;
            if (accum) y[oix] += outv; else y[oix] = outv;
        }
        __syncthreads();
    }
}

// ---------------------------------------------------------------------------
extern "C" void kernel_launch(void* const* d_in, const int* in_sizes, int n_in,
                              void* d_out, int out_size, void* d_ws, size_t ws_size,
                              hipStream_t stream) {
    (void)in_sizes; (void)n_in; (void)out_size; (void)ws_size;

    // ---- inputs in setup_inputs() dict order ----
    const float* x         = (const float*)d_in[0];
    const float* patch_w   = (const float*)d_in[1];   // (384, 768) flat
    const float* patch_b   = (const float*)d_in[2];
    const float* cls_tok   = (const float*)d_in[3];
    const float* pos_emb   = (const float*)d_in[4];
    const float* normf     = (const float*)d_in[5];
    const float* sp_conv_w = (const float*)d_in[6];
    const float* sp_conv_b = (const float*)d_in[7];
    const float* sp_xproj  = (const float*)d_in[8];
    const float* sp_dtw    = (const float*)d_in[9];
    const float* sp_dtb    = (const float*)d_in[10];
    const float* sp_Alog   = (const float*)d_in[11];
    const float* sp_Dp     = (const float*)d_in[12];
    const float* sp_conv_w_r = (const float*)d_in[13];
    const float* sp_conv_b_r = (const float*)d_in[14];
    const float* sp_xproj_r  = (const float*)d_in[15];
    const float* sp_dtw_r    = (const float*)d_in[16];
    const float* sp_dtb_r    = (const float*)d_in[17];
    const float* sp_Alog_r   = (const float*)d_in[18];
    const float* sp_Dp_r     = (const float*)d_in[19];
    const float* tm_conv_w = (const float*)d_in[20];
    const float* tm_conv_b = (const float*)d_in[21];
    const float* tm_xproj  = (const float*)d_in[22];
    const float* tm_dtw    = (const float*)d_in[23];
    const float* tm_dtb    = (const float*)d_in[24];
    const float* tm_Alog   = (const float*)d_in[25];
    const float* tm_Dp     = (const float*)d_in[26];
    const float* sp_norm    = (const float*)d_in[27];
    const float* sp_inproj  = (const float*)d_in[28];
    const float* sp_outproj = (const float*)d_in[29];
    const float* tm_norm    = (const float*)d_in[30];
    const float* tm_inproj  = (const float*)d_in[31];
    const float* tm_outproj = (const float*)d_in[32];

    // ---- workspace arena ----
    size_t off = 0;
    auto arena = [&](size_t bytes) {
        void* p = (char*)d_ws + off;
        off += (bytes + 255) & ~(size_t)255;
        return p;
    };
    float* hidden   = (float*)arena((size_t)NTOK * DMODEL * 4);
    float* residual = (float*)arena((size_t)NTOK * DMODEL * 4);
    float* normed   = (float*)arena((size_t)NTOK * DMODEL * 4);
    float* xz       = (float*)arena((size_t)NTOK * 2 * EMODEL * 4);
    float* im2col   = xz;                                  // alias (used only before xz)
    float* u_act    = (float*)arena((size_t)NTOK * EMODEL * 4);
    float* ptokbuf  = u_act;                               // alias (used only before u_act)
    float* dblbuf   = (float*)arena((size_t)NTOK * 64 * 4);  // 56 cols, padded ld=64
    float* dtbuf    = (float*)arena((size_t)NTOK * EMODEL * 4);
    float* ybuf     = (float*)arena((size_t)NTOK * EMODEL * 4);
    float* dtw_pad  = (float*)arena((size_t)EMODEL * 32 * 4);

    auto gemm = [&](const float* A, int lda, const float* W, int ldw, const float* bias,
                    float* out, int ldo, int M, int N, int K, int act) {
        dim3 blk(32, 4);
        dim3 grd((M + 15) / 16, (N + 127) / 128);
        gemm_wmma_kernel<<<grd, blk, 0, stream>>>(A, lda, W, ldw, bias, out, ldo, M, N, K, act);
    };

    // ---- patch embedding: im2col + GEMM + assemble ----
    im2col_patch_kernel<<<PTOK, 256, 0, stream>>>(x, im2col);
    gemm(im2col, 768, patch_w, 768, patch_b, ptokbuf, DMODEL, PTOK, DMODEL, 768, 0);
    assemble_kernel<<<NTOK, 128, 0, stream>>>(ptokbuf, cls_tok, pos_emb, hidden, residual);

    auto branch = [&](const float* cw, const float* cb, const float* xp, const float* dw,
                      const float* db, const float* al, const float* dp,
                      int L, int Nseq, int dir, int accum) {
        conv_silu_kernel<<<NTOK, 256, 0, stream>>>(xz, cw, cb, u_act, L, dir);
        gemm(u_act, EMODEL, xp, EMODEL, nullptr, dblbuf, 64, NTOK, RRANK + 2 * SSTATE, EMODEL, 0);
        pad_dtw_kernel<<<EMODEL, 32, 0, stream>>>(dw, dtw_pad);
        // K=32 with zero-padded weights; dbl cols 24..31 (B values) hit zero weights.
        gemm(dblbuf, 64, dtw_pad, 32, db, dtbuf, EMODEL, NTOK, EMODEL, 32, 1);
        scan_kernel<<<Nseq, 256, 0, stream>>>(dblbuf, dtbuf, u_act, xz, al, dp, ybuf, L, dir, accum);
    };

    // ---- 12 spatial bi-mamba blocks: 16 sequences of L=197 ----
    for (int i = 0; i < 12; ++i) {
        resid_rmsnorm_kernel<<<NTOK, 128, 0, stream>>>(residual, hidden, sp_norm + (size_t)i * DMODEL, normed);
        gemm(normed, DMODEL, sp_inproj + (size_t)i * 2 * EMODEL * DMODEL, DMODEL, nullptr,
             xz, 2 * EMODEL, NTOK, 2 * EMODEL, DMODEL, 0);
        branch(sp_conv_w + (size_t)i * EMODEL * 4, sp_conv_b + (size_t)i * EMODEL,
               sp_xproj + (size_t)i * 56 * EMODEL, sp_dtw + (size_t)i * EMODEL * RRANK,
               sp_dtb + (size_t)i * EMODEL, sp_Alog + (size_t)i * EMODEL * SSTATE,
               sp_Dp + (size_t)i * EMODEL, 197, 16, /*dir=*/0, /*accum=*/0);
        branch(sp_conv_w_r + (size_t)i * EMODEL * 4, sp_conv_b_r + (size_t)i * EMODEL,
               sp_xproj_r + (size_t)i * 56 * EMODEL, sp_dtw_r + (size_t)i * EMODEL * RRANK,
               sp_dtb_r + (size_t)i * EMODEL, sp_Alog_r + (size_t)i * EMODEL * SSTATE,
               sp_Dp_r + (size_t)i * EMODEL, 197, 16, /*dir=*/1, /*accum=*/1);
        gemm(ybuf, EMODEL, sp_outproj + (size_t)i * DMODEL * EMODEL, EMODEL, nullptr,
             hidden, DMODEL, NTOK, DMODEL, EMODEL, 0);
    }

    // ---- 12 temporal mamba blocks: 2 sequences of L=1576 ----
    for (int i = 0; i < 12; ++i) {
        resid_rmsnorm_kernel<<<NTOK, 128, 0, stream>>>(residual, hidden, tm_norm + (size_t)i * DMODEL, normed);
        gemm(normed, DMODEL, tm_inproj + (size_t)i * 2 * EMODEL * DMODEL, DMODEL, nullptr,
             xz, 2 * EMODEL, NTOK, 2 * EMODEL, DMODEL, 0);
        branch(tm_conv_w + (size_t)i * EMODEL * 4, tm_conv_b + (size_t)i * EMODEL,
               tm_xproj + (size_t)i * 56 * EMODEL, tm_dtw + (size_t)i * EMODEL * RRANK,
               tm_dtb + (size_t)i * EMODEL, tm_Alog + (size_t)i * EMODEL * SSTATE,
               tm_Dp + (size_t)i * EMODEL, 1576, 2, /*dir=*/0, /*accum=*/0);
        gemm(ybuf, EMODEL, tm_outproj + (size_t)i * DMODEL * EMODEL, EMODEL, nullptr,
             hidden, DMODEL, NTOK, DMODEL, EMODEL, 0);
    }

    // ---- final rmsnorm(hidden + residual) -> d_out ----
    resid_rmsnorm_kernel<<<NTOK, 128, 0, stream>>>(residual, hidden, normf, (float*)d_out);
}
